// CFConv_68478958567674
// MI455X (gfx1250) — matrix operand order repaired and checked by reading, not compiled, not run on hardware
//
#include <hip/hip_runtime.h>
#include <hip/hip_bf16.h>

// ---------------------------------------------------------------------------
// CFConv fused kernel for gfx1250 (MI455X).
//   f1 = ssp(r @ W1 + b1)          (per (b,i): 256x64 @ 64x128)
//   f2 = ssp(f1 @ W2 + b2)         (256x128 @ 128x128)
//   out[b,i,:] = sum_j x[b,j,:] * f2[j,:]
// One workgroup per (b,i), 8 waves. bf16 WMMA, f32 accumulate.
// Weights pre-swizzled in LDS into B-fragment layout (one 32B per-lane load);
// staging fully vectorized; ssp computed with raw v_exp_f32/v_log_f32 (base-2).
// ---------------------------------------------------------------------------

typedef __bf16        v16bf  __attribute__((ext_vector_type(16)));
typedef float         v8f    __attribute__((ext_vector_type(8)));
typedef unsigned int  v8u    __attribute__((ext_vector_type(8)));   // 32B
typedef unsigned int  u32x4  __attribute__((ext_vector_type(4)));   // 16B
typedef unsigned int  u32x2  __attribute__((ext_vector_type(2)));   // 8B
typedef float         f32x4  __attribute__((ext_vector_type(4)));   // 16B

#define BN    256   // N
#define DM    128   // d
#define RBF   64
#define RS    72    // LDS row stride (elems) for r tile (256 x 64)
#define L1S   136   // LDS row stride for layer-1 activations (16 x 128 per wave)

#define LN2    0.69314718056f
#define LOG2E  1.44269504089f

__device__ __forceinline__ unsigned short f2bf(float f) {
    union { float f; unsigned int u; } v; v.f = f;
    unsigned int r = v.u + 0x7FFFu + ((v.u >> 16) & 1u);   // round-nearest-even
    return (unsigned short)(r >> 16);
}

// shifted softplus ln(1+e^x) - ln2, computed base-2 with raw HW transcendentals:
//   max(x,0) + ln2 * log2(1 + exp2(-|x|*log2e)) - ln2
// exp2 arg <= 0 and log2 arg in [1,2] -> bare v_exp_f32 / v_log_f32 are safe.
__device__ __forceinline__ float ssp(float v) {
    float m = fmaxf(v, 0.0f);
    float e = __builtin_amdgcn_exp2f(-LOG2E * fabsf(v));   // (0, 1]
    float l = __builtin_amdgcn_logf(1.0f + e);             // [0, 1]
    return fmaf(LN2, l, m) - LN2;
}

// A-fragment (16x32 bf16): lane's row = lane&15.
// p = &row[k0], k0 = ks*32 + (lane>>4)*8 ; reads K {k0..k0+7, k0+16..k0+23}.
__device__ __forceinline__ v16bf loadA(const unsigned short* p) {
    const u32x4* q0 = (const u32x4*)__builtin_assume_aligned(p, 16);
    const u32x4* q1 = (const u32x4*)__builtin_assume_aligned(p + 16, 16);
    union { u32x4 q[2]; v16bf b; } u;
    u.q[0] = *q0;
    u.q[1] = *q1;
    return u.b;
}

// B-fragment from pre-swizzled LDS: frag #f, this lane's 16 bf16 = 32 contiguous B.
__device__ __forceinline__ v16bf loadBfrag(const unsigned short* fragBase,
                                           int f, int lane) {
    const v8u* p = (const v8u*)__builtin_assume_aligned(
        fragBase + (size_t)((f << 5) | lane) * 16, 32);
    union { v8u u; v16bf b; } r;
    r.u = *p;
    return r.b;
}

// Stage a KxN (N=128) f32 weight matrix into B-fragment-swizzled bf16 LDS.
// Fragment (ks,t): K rows [ks*32,+32), cols [t*16,+16); within a fragment:
// lane = nrel | (((krel>>3)&1)<<4), packed dword slot = ((krel>>4)<<2) | ((krel&7)>>1).
// Process 2 K-rows x 4 cols per step: each packed dword is one frag dword.
template<int K>
__device__ __forceinline__ void stageBswz(unsigned short* dst,
                                          const float* __restrict__ src,
                                          int tid) {
    unsigned int* d32 = (unsigned int*)dst;
    const f32x4* s4 = (const f32x4*)src;
#pragma unroll
    for (int c = 0; c < (K * DM) / (8 * 256); ++c) {
        int e  = tid + c * 256;          // chunk: 2 rows x 4 cols
        int nq = e & 31;                 // column quad (128/4)
        int kp = e >> 5;                 // K row-pair
        f32x4 w0 = s4[(2 * kp) * 32 + nq];       // global_load_b128
        f32x4 w1 = s4[(2 * kp + 1) * 32 + nq];   // global_load_b128
        int k    = 2 * kp;
        int ks   = k >> 5;
        int hi   = (k >> 3) & 1;
        int slot = (((k >> 4) & 1) << 2) | (kp & 3);   // dword slot in lane's 8 dwords
#pragma unroll
        for (int q = 0; q < 4; ++q) {
            int n = nq * 4 + q;
            int t = n >> 4, nr = n & 15;
            int lane = nr | (hi << 4);
            unsigned int u = (unsigned)f2bf(w0[q]) | ((unsigned)f2bf(w1[q]) << 16);
            d32[(size_t)((((ks << 3) | t) << 5) | lane) * 8 + slot] = u;
        }
    }
}

__global__ __launch_bounds__(256)
void cfconv_fused_kernel(const float* __restrict__ x,   // (B,N,d)
                         const float* __restrict__ r,   // (B,N,N,rbf)
                         const float* __restrict__ W1,  // (rbf,d)
                         const float* __restrict__ b1,  // (d)
                         const float* __restrict__ W2,  // (d,d)
                         const float* __restrict__ b2,  // (d)
                         float* __restrict__ out)       // (B,N,d)
{
    __shared__ __attribute__((aligned(32))) unsigned short sW1[RBF * DM];     // 16 KB
    __shared__ __attribute__((aligned(32))) unsigned short sW2[DM * DM];      // 32 KB
    __shared__ __attribute__((aligned(16))) unsigned short sR [BN * RS];      // 36 KB
    __shared__ __attribute__((aligned(16))) unsigned short sL1[8 * 16 * L1S]; // 34 KB
    __shared__ float sOut[DM];

    const int tid  = threadIdx.x;
    const int lane = tid & 31;
    const int wv   = tid >> 5;       // wave id 0..7
    const int half = lane >> 4;      // 0: lanes 0-15, 1: lanes 16-31
    const int lm   = lane & 15;

    const int bid = blockIdx.x;      // b*256 + i
    const int b   = bid >> 8;

    // ---- biases into registers (broadcast loads, L2 hit) -------------------
    float bias1[8], bias2[8];
#pragma unroll
    for (int t = 0; t < 8; ++t) {
        bias1[t] = b1[t * 16 + lm];
        bias2[t] = b2[t * 16 + lm];
    }

    // ---- stage weights (f32 -> bf16, B-fragment swizzle) -------------------
    stageBswz<RBF>(sW1, W1, tid);
    stageBswz<DM >(sW2, W2, tid);
    if (tid < DM) sOut[tid] = 0.0f;

    // ---- stage r[b,i,:,:] (256 x 64 f32 -> bf16, row-major) ----------------
    const float* rbase = r + (size_t)bid * (BN * RBF);
    const f32x4* r4 = (const f32x4*)rbase;
#pragma unroll
    for (int c = 0; c < (BN * RBF) / (4 * 256); ++c) {    // 16 x b128 per thread
        int e  = tid + c * 256;          // float4 index
        int j  = e >> 4;                 // 16 quads per 64-wide row
        int kq = e & 15;
        f32x4 w = r4[e];
        u32x2 p;
        p[0] = (unsigned)f2bf(w[0]) | ((unsigned)f2bf(w[1]) << 16);
        p[1] = (unsigned)f2bf(w[2]) | ((unsigned)f2bf(w[3]) << 16);
        *(u32x2*)__builtin_assume_aligned(&sR[j * RS + kq * 4], 8) = p;
    }
    __syncthreads();

    const float* xb = x + (size_t)b * (BN * DM);
    float accOut[8];
#pragma unroll
    for (int t = 0; t < 8; ++t) accOut[t] = 0.0f;

    unsigned short* myL1 = &sL1[wv * 16 * L1S];

    // ---- two M-iterations: wave handles j rows [mi*128 + wv*16, +16) -------
    // sL1 strip is produced and consumed by the same wave (LDS ops are
    // in-order per wave), so no workgroup barriers inside this loop.
    for (int mi = 0; mi < 2; ++mi) {
        const int jb = mi * 128 + wv * 16;

        // ===== layer 1: (16 x 64) @ (64 x 128), K=64 -> 2 k-steps ==========
        v16bf a1[2];
#pragma unroll
        for (int ks = 0; ks < 2; ++ks)
            a1[ks] = loadA(&sR[(size_t)(jb + lm) * RS + ks * 32 + half * 8]);

#pragma unroll
        for (int t = 0; t < 8; ++t) {
            v8f c = {};
#pragma unroll
            for (int ks = 0; ks < 2; ++ks) {
                v16bf bf = loadBfrag(sW1, (ks << 3) | t, lane);
                c = __builtin_amdgcn_wmma_f32_16x16x32_bf16(
                        false, a1[ks], false, bf, (short)0, c, false, false);
            }
            const int col = t * 16 + lm;
#pragma unroll
            for (int v = 0; v < 8; ++v) {
                int m = v + half * 8;
                myL1[m * L1S + col] = f2bf(ssp(c[v] + bias1[t]));
            }
        }

        // ===== layer 2: (16 x 128) @ (128 x 128), K=128 -> 4 k-steps =======
        v16bf a2[4];
#pragma unroll
        for (int ks = 0; ks < 4; ++ks)
            a2[ks] = loadA(&myL1[lm * L1S + ks * 32 + half * 8]);

#pragma unroll
        for (int t = 0; t < 8; ++t) {
            v8f c = {};
#pragma unroll
            for (int ks = 0; ks < 4; ++ks) {
                v16bf bf = loadBfrag(sW2, (ks << 3) | t, lane);
                c = __builtin_amdgcn_wmma_f32_16x16x32_bf16(
                        false, a2[ks], false, bf, (short)0, c, false, false);
            }
            // fused epilogue: ssp + multiply by x[b,j,col], accumulate over j
            const int col = t * 16 + lm;
            const float* xcol = xb + (size_t)(jb + half * 8) * DM + col;
#pragma unroll
            for (int v = 0; v < 8; ++v) {
                float y = ssp(c[v] + bias2[t]);
                accOut[t] += y * xcol[v * DM];
            }
        }
    }

    // ---- reduce over lane halves, then across waves ------------------------
#pragma unroll
    for (int t = 0; t < 8; ++t) {
        accOut[t] += __shfl_xor(accOut[t], 16, 32);
        if (half == 0) atomicAdd(&sOut[t * 16 + lm], accOut[t]);  // ds_add_f32
    }
    __syncthreads();

    if (tid < DM) out[(size_t)bid * DM + tid] = sOut[tid];
}

extern "C" void kernel_launch(void* const* d_in, const int* in_sizes, int n_in,
                              void* d_out, int out_size, void* d_ws, size_t ws_size,
                              hipStream_t stream) {
    const float* x  = (const float*)d_in[0];
    const float* r  = (const float*)d_in[1];
    const float* W1 = (const float*)d_in[2];
    const float* b1 = (const float*)d_in[3];
    const float* W2 = (const float*)d_in[4];
    const float* b2 = (const float*)d_in[5];
    float* out = (float*)d_out;

    // One workgroup per (b, i): 8 * 256 = 2048 blocks of 256 threads (8 waves).
    cfconv_fused_kernel<<<dim3(8 * 256), dim3(256), 0, stream>>>(
        x, r, W1, b1, W2, b2, out);
}
